// GeometricBlock_73469710566102
// MI455X (gfx1250) — matrix operand order repaired and checked by reading, not compiled, not run on hardware
//
#include <hip/hip_runtime.h>
#include <hip/hip_bf16.h>

typedef __attribute__((ext_vector_type(16))) __bf16 v16bf;
typedef __attribute__((ext_vector_type(8)))  __bf16 v8bf;
typedef __attribute__((ext_vector_type(4)))  __bf16 v4bf;
typedef __attribute__((ext_vector_type(8)))  float  v8f;

#define N_NODES   20000
#define N_EDGES   640000
#define DIM       128
#define NTILES    (N_EDGES / 16)
#define QK_TILES  (N_NODES / 16)
#define WAVES     8

// Per-wave LDS region layout (bytes) in the edge kernel:
//   0     raw_rad  f32[16][64]   4096   (async-load target)
//   4096  raw_sph  f32[16][64]   4096   (async-load target)
//   8192  sin_rad  bf16[16][64]  2048
//   10240 sin_sph  bf16[16][64]  2048
//   12288 sh1      bf16[16][128] 4096
//   16384 sw       f32[16][128]  8192
#define WAVE_LDS     24576
#define WEIGHT_BYTES (2 * (64 * 128 + 128 * 128) * 2)   // 98304

__device__ __forceinline__ __bf16 to_bf(float x) { return (__bf16)x; }

__device__ __forceinline__ v8f vzero8() {
  v8f z = {0.f, 0.f, 0.f, 0.f, 0.f, 0.f, 0.f, 0.f};
  return z;
}

// Assemble a 16-element bf16 fragment from two contiguous 128-bit LDS loads.
__device__ __forceinline__ v16bf ld_frag(const __bf16* p, int gap) {
  v8bf lo = *(const v8bf*)p;
  v8bf hi = *(const v8bf*)(p + gap);
  return __builtin_shufflevector(lo, hi, 0, 1, 2, 3, 4, 5, 6, 7,
                                         8, 9, 10, 11, 12, 13, 14, 15);
}

// One 16x16x32 bf16 WMMA step, f32 accumulate.
//  A : row-major [16 x lda] bf16 tile in LDS (lda multiple of 8)
//  BT: column-major weights in LDS, BT[col*ldk + k]
__device__ __forceinline__ v8f wmma_step(const __bf16* A, int lda, int K0,
                                         const __bf16* BT, int ldk, int colbase,
                                         v8f acc) {
  const int lane = threadIdx.x & 31;
  const int row  = lane & 15;
  const int hs   = lane >> 4;
  v16bf a = ld_frag(A + row * lda + K0 + hs * 8, 16);
  v16bf b = ld_frag(BT + (colbase + row) * ldk + K0 + hs * 8, 16);
  return __builtin_amdgcn_wmma_f32_16x16x32_bf16(false, a, false, b,
                                                 (short)0, acc, false, false);
}

// Async-copy a 16x64 f32 tile (4 KB) global -> LDS, tracked by ASYNCcnt.
// Flat LDS aperture: low 32 bits of a generic LDS pointer == LDS byte offset.
__device__ __forceinline__ void prefetch_tile_async(const float* __restrict__ g,
                                                    const float* lds_dst) {
  const int lane = threadIdx.x & 31;
#pragma unroll
  for (int i = 0; i < 8; ++i) {
    const float* gp = g + i * 128 + lane * 4;                 // 16 B per lane
    unsigned dst = (unsigned)(size_t)(lds_dst + i * 128 + lane * 4);
    asm volatile("global_load_async_to_lds_b128 %0, %1, off"
                 :: "v"(dst), "v"(gp) : "memory");
  }
}

__device__ __forceinline__ void wait_async0() {
  asm volatile("s_wait_asynccnt 0x0" ::: "memory");
}

// Convert a raw f32 [16][64] LDS tile into a bf16 [16][64] LDS tile.
__device__ __forceinline__ void convert_tile(const float* __restrict__ raw,
                                             __bf16* __restrict__ dst) {
  const int lane = threadIdx.x & 31;
#pragma unroll
  for (int i = 0; i < 8; ++i) {
    float4 v = *(const float4*)(raw + i * 128 + lane * 4);
    v4bf b = {to_bf(v.x), to_bf(v.y), to_bf(v.z), to_bf(v.w)};
    *(v4bf*)(dst + i * 128 + lane * 4) = b;
  }
}

// ---------------------------------------------------------------------------
// Kernel 1: Q = node_feats @ Wq, K = node_feats @ Wk  (grid.y: 0=Q, 1=K)
// ---------------------------------------------------------------------------
__global__ __launch_bounds__(256)
void qk_project(const float* __restrict__ node_feats,
                const float* __restrict__ Wq, const float* __restrict__ Wk,
                float* __restrict__ Qout, float* __restrict__ Kout) {
  __shared__ __bf16 sA[16 * DIM];        // row-major node tile
  __shared__ __bf16 sWT[DIM * DIM];      // transposed weights [col][k]
  const float* W = blockIdx.y ? Wk : Wq;
  float* Out     = blockIdx.y ? Kout : Qout;
  const int t    = threadIdx.x;
  const int tile = blockIdx.x;

#pragma unroll
  for (int i = 0; i < 8; ++i) {
    int e = t * 8 + i;
    sA[e] = to_bf(node_feats[(size_t)tile * (16 * DIM) + e]);
  }
#pragma unroll 4
  for (int i = 0; i < 64; ++i) {
    int e = i * 256 + t;
    sWT[(e & 127) * DIM + (e >> 7)] = to_bf(W[e]);
  }
  __syncthreads();

  const int wv = t >> 5, lane = t & 31;
  v8f acc = vzero8();
#pragma unroll
  for (int kk = 0; kk < 4; ++kk)
    acc = wmma_step(sA, DIM, kk * 32, sWT, DIM, wv * 16, acc);

  const int hs  = lane >> 4;
  const int col = wv * 16 + (lane & 15);
#pragma unroll
  for (int g = 0; g < 8; ++g) {
    int M = g + hs * 8;
    Out[(size_t)(tile * 16 + M) * DIM + col] = acc[g];
  }
}

// ---------------------------------------------------------------------------
// Kernel 2: fused edge-filter MLPs (bf16 WMMA, async double-buffered inputs)
// + filter-scaled attention + irrep broadcast + atomic scatter.
// ---------------------------------------------------------------------------
__global__ __launch_bounds__(256)
void edge_attention(const float* __restrict__ edge_sh,
                    const float* __restrict__ edge_feats,
                    const float* __restrict__ chi_scalar,
                    const float* __restrict__ cutoffs,
                    const int*   __restrict__ senders,
                    const int*   __restrict__ receivers,
                    const float* __restrict__ W_rad1, const float* __restrict__ b_rad1,
                    const float* __restrict__ W_rad2, const float* __restrict__ b_rad2,
                    const float* __restrict__ W_sph1, const float* __restrict__ b_sph1,
                    const float* __restrict__ W_sph2, const float* __restrict__ b_sph2,
                    const float* __restrict__ Qn, const float* __restrict__ Kn,
                    float* __restrict__ out) {
  extern __shared__ char smem[];
  __bf16* sWr1 = (__bf16*)smem;            // [128 col][64 k]   16 KB (transposed)
  __bf16* sWr2 = sWr1 + 64 * 128;          // [128 col][128 k]  32 KB
  __bf16* sWs1 = sWr2 + 128 * 128;         //                   16 KB
  __bf16* sWs2 = sWs1 + 64 * 128;          //                   32 KB
  char* wbase  = smem + WEIGHT_BYTES;

  const int t = threadIdx.x, wv = t >> 5, lane = t & 31;
  char* my = wbase + wv * WAVE_LDS;
  float*  raw_rad = (float*) (my);
  float*  raw_sph = (float*) (my + 4096);
  __bf16* sin_rad = (__bf16*)(my + 8192);
  __bf16* sin_sph = (__bf16*)(my + 10240);
  __bf16* sh1     = (__bf16*)(my + 12288);
  float*  sw      = (float*) (my + 16384);

  // Stage all four weight matrices, transposed, bf16 (once per block).
#pragma unroll 2
  for (int i = 0; i < 32; ++i) { int e = i * 256 + t; sWr1[(e & 127) * 64  + (e >> 7)] = to_bf(W_rad1[e]); }
#pragma unroll 2
  for (int i = 0; i < 64; ++i) { int e = i * 256 + t; sWr2[(e & 127) * 128 + (e >> 7)] = to_bf(W_rad2[e]); }
#pragma unroll 2
  for (int i = 0; i < 32; ++i) { int e = i * 256 + t; sWs1[(e & 127) * 64  + (e >> 7)] = to_bf(W_sph1[e]); }
#pragma unroll 2
  for (int i = 0; i < 64; ++i) { int e = i * 256 + t; sWs2[(e & 127) * 128 + (e >> 7)] = to_bf(W_sph2[e]); }
  __syncthreads();   // only barrier; everything below is wave-local

  const int row = lane & 15, hs = lane >> 4;
  const int tile0  = blockIdx.x * WAVES + wv;
  const int stride = gridDim.x * WAVES;

  // Prime the async pipeline with this wave's first tile.
  if (tile0 < NTILES) {
    prefetch_tile_async(edge_feats + (size_t)tile0 * (16 * 64), raw_rad);
    prefetch_tile_async(chi_scalar + (size_t)tile0 * (16 * 64), raw_sph);
  }

  for (int tile = tile0; tile < NTILES; tile += stride) {
    const int e0 = tile * 16;

    // Land the async tiles, convert to bf16, then immediately issue the
    // next tile's async loads so global traffic overlaps the WMMAs below.
    wait_async0();
    convert_tile(raw_rad, sin_rad);
    convert_tile(raw_sph, sin_sph);
    const int nxt = tile + stride;
    if (nxt < NTILES) {
      prefetch_tile_async(edge_feats + (size_t)nxt * (16 * 64), raw_rad);
      prefetch_tile_async(chi_scalar + (size_t)nxt * (16 * 64), raw_sph);
    }

    v8f wacc[8];
#pragma unroll
    for (int c = 0; c < 8; ++c) wacc[c] = vzero8();

#pragma unroll
    for (int br = 0; br < 2; ++br) {
      const __bf16* sin = br ? sin_sph : sin_rad;
      const __bf16* W1  = br ? sWs1 : sWr1;
      const __bf16* W2  = br ? sWs2 : sWr2;
      const float*  b1  = br ? b_sph1 : b_rad1;

      // GEMM1: [16x64]@[64x128] + bias, SiLU -> sh1 (bf16)
#pragma unroll
      for (int ct = 0; ct < 8; ++ct) {
        v8f acc = vzero8();
        acc = wmma_step(sin, 64, 0,  W1, 64, ct * 16, acc);
        acc = wmma_step(sin, 64, 32, W1, 64, ct * 16, acc);
        const int col = ct * 16 + row;
        const float bb = b1[col];
#pragma unroll
        for (int g = 0; g < 8; ++g) {
          float x  = acc[g] + bb;
          float sx = x / (1.0f + __expf(-x));
          sh1[(g + hs * 8) * 128 + col] = to_bf(sx);
        }
      }
      // GEMM2: [16x128]@[128x128], accumulate into wacc
#pragma unroll
      for (int kk = 0; kk < 4; ++kk) {
#pragma unroll
        for (int ct = 0; ct < 8; ++ct)
          wacc[ct] = wmma_step(sh1, 128, kk * 32, W2, 128, ct * 16, wacc[ct]);
      }
    }

    // Output biases, park filter w in LDS as f32 [16 edges][128].
#pragma unroll
    for (int ct = 0; ct < 8; ++ct) {
      const int col = ct * 16 + row;
      const float bb = b_rad2[col] + b_sph2[col];
#pragma unroll
      for (int g = 0; g < 8; ++g)
        sw[(g + hs * 8) * 128 + col] = wacc[ct][g] + bb;
    }

    // Attention: lane = (edge row, half). half covers 64 dims = 2 heads.
    const int ge = e0 + row;
    const int r  = receivers[ge];
    const int s  = senders[ge];
    const float4* qv = (const float4*)(Qn + (size_t)r * DIM + hs * 64);
    const float4* kv = (const float4*)(Kn + (size_t)s * DIM + hs * 64);
    const float*  wp = sw + row * DIM + hs * 64;
    float aA = 0.f, aB = 0.f;
#pragma unroll
    for (int i = 0; i < 16; ++i) {
      float4 q4 = qv[i], k4 = kv[i];
      float su = q4.x * k4.x * wp[i * 4 + 0] + q4.y * k4.y * wp[i * 4 + 1] +
                 q4.z * k4.z * wp[i * 4 + 2] + q4.w * k4.w * wp[i * 4 + 3];
      if (i < 8) aA += su; else aB += su;
    }
    const float sc = cutoffs[ge] * 0.00552427172801991f;  // 1/(sqrt(32)*32)
    aA *= sc; aB *= sc;
    const float* shp = edge_sh + (size_t)ge * 16;
    float* op = out + (size_t)r * 16;
    if (hs == 0) {  // heads 0,1 -> irreps of dim 1 and 3
      atomicAdd(op + 0, aA * shp[0]);
#pragma unroll
      for (int m = 0; m < 3; ++m) atomicAdd(op + 1 + m, aB * shp[1 + m]);
    } else {        // heads 2,3 -> irreps of dim 5 and 7
#pragma unroll
      for (int m = 0; m < 5; ++m) atomicAdd(op + 4 + m, aA * shp[4 + m]);
#pragma unroll
      for (int m = 0; m < 7; ++m) atomicAdd(op + 9 + m, aB * shp[9 + m]);
    }
  }
}

// ---------------------------------------------------------------------------
extern "C" void kernel_launch(void* const* d_in, const int* in_sizes, int n_in,
                              void* d_out, int out_size, void* d_ws, size_t ws_size,
                              hipStream_t stream) {
  const float* edge_sh    = (const float*)d_in[0];
  const float* node_feats = (const float*)d_in[1];
  const float* edge_feats = (const float*)d_in[2];
  const float* chi_scalar = (const float*)d_in[3];
  const float* cutoffs    = (const float*)d_in[4];
  const int*   senders    = (const int*)d_in[5];
  const int*   receivers  = (const int*)d_in[6];
  const float* W_rad1 = (const float*)d_in[7];
  const float* b_rad1 = (const float*)d_in[8];
  const float* W_rad2 = (const float*)d_in[9];
  const float* b_rad2 = (const float*)d_in[10];
  const float* W_sph1 = (const float*)d_in[11];
  const float* b_sph1 = (const float*)d_in[12];
  const float* W_sph2 = (const float*)d_in[13];
  const float* b_sph2 = (const float*)d_in[14];
  const float* Wq = (const float*)d_in[15];
  const float* Wk = (const float*)d_in[16];
  float* out = (float*)d_out;

  float* Q = (float*)d_ws;
  float* K = Q + (size_t)N_NODES * DIM;

  hipMemsetAsync(out, 0, (size_t)out_size * sizeof(float), stream);

  qk_project<<<dim3(QK_TILES, 2), 256, 0, stream>>>(node_feats, Wq, Wk, Q, K);

  const size_t lds = WEIGHT_BYTES + (size_t)WAVES * WAVE_LDS;  // 294912 B
  edge_attention<<<dim3(1024), 256, lds, stream>>>(
      edge_sh, edge_feats, chi_scalar, cutoffs, senders, receivers,
      W_rad1, b_rad1, W_rad2, b_rad2, W_sph1, b_sph1, W_sph2, b_sph2,
      Q, K, out);
}
